// CascadeRCNN_83854941487832
// MI455X (gfx1250) — matrix operand order repaired and checked by reading, not compiled, not run on hardware
//
#include <hip/hip_runtime.h>
#include <hip/hip_bf16.h>

typedef __attribute__((ext_vector_type(16))) __bf16 v16bf;
typedef __attribute__((ext_vector_type(8)))  float  v8f;

#define NROI  1000
#define MPAD  1024
#define K1    12544     // 7*7*256
#define NH    1024
#define NCLS  81

__device__ __forceinline__ unsigned short f2bf(float x) {
  unsigned u = __float_as_uint(x);
  u += 0x7FFFu + ((u >> 16) & 1u);          // round-to-nearest-even
  return (unsigned short)(u >> 16);
}

// ---------------------------------------------------------------- utilities
__global__ void zero_f32_kernel(float* __restrict__ p, int n) {
  int i = blockIdx.x * 256 + threadIdx.x;
  if (i < n) p[i] = 0.0f;
}
__global__ void zero_u16_kernel(unsigned short* __restrict__ p, int n) {
  int i = blockIdx.x * 256 + threadIdx.x;
  if (i < n) p[i] = 0;
}

// ------------------------------------------------- weight pack fp32 -> bf16
// W: [K][N] row-major fp32  ->  Bp: [K/32][N][32] bf16
// Lane layout for V_WMMA_*_16x16x32_bf16 B operand: lane L reads halves
// h = (L>>4)*8 .. +8 and (L>>4)*8+16 .. +8 of column n = n0 + (L&15): two 16B chunks.
__global__ __launch_bounds__(256) void pack_w_kernel(
    const float* __restrict__ W, unsigned short* __restrict__ Bp, int K, int N) {
  int nb   = N >> 8;                       // n-blocks of 256
  int kb   = blockIdx.x / nb;
  int nblk = blockIdx.x - kb * nb;
  int n    = nblk * 256 + threadIdx.x;
  alignas(16) unsigned short tmp[32];
#pragma unroll
  for (int h = 0; h < 32; ++h) {           // coalesced reads across threads
    tmp[h] = f2bf(W[(size_t)(kb * 32 + h) * N + n]);
  }
  unsigned short* dst = Bp + ((size_t)kb * N + n) * 32;
#pragma unroll
  for (int i = 0; i < 4; ++i)              // 64B contiguous store per thread
    ((float4*)dst)[i] = ((const float4*)tmp)[i];
}

// ------------------------------------------------------- pyramid ROI align
// One block (256 ch) per (roi, py, px). Writes bf16 A row-major [MPAD][K1].
__global__ __launch_bounds__(256) void roi_align_kernel(
    const float* __restrict__ P2, const float* __restrict__ P3,
    const float* __restrict__ P4, const float* __restrict__ P5,
    const float* __restrict__ rois, unsigned short* __restrict__ A1) {
  int blk = blockIdx.x;                    // roi*49 + pos
  int roi = blk / 49;
  int pos = blk - roi * 49;
  int py  = pos / 7, px = pos - py * 7;
  int c   = threadIdx.x;

  float y1 = rois[roi * 4 + 0], x1 = rois[roi * 4 + 1];
  float y2 = rois[roi * 4 + 2], x2 = rois[roi * 4 + 3];
  float h = y2 - y1, w = x2 - x1;
  float lvlf = floorf(4.0f + log2f(sqrtf(fmaxf(h * w, 1e-6f)) / 224.0f));
  lvlf = fminf(fmaxf(lvlf, 2.0f), 5.0f);
  int lev = (int)lvlf - 2;                 // 0..3

  const float* feat = P2;
  if (lev == 1) feat = P3; else if (lev == 2) feat = P4; else if (lev == 3) feat = P5;
  int   H   = 256 >> lev;                  // square levels: 256,128,64,32
  float inv = 1.0f / (float)(4 << lev);

  float sy1 = y1 * inv, sx1 = x1 * inv, sy2 = y2 * inv, sx2 = x2 * inv;
  float gy = ((float)py + 0.5f) / 7.0f;
  float gx = ((float)px + 0.5f) / 7.0f;
  float ys = sy1 + gy * (sy2 - sy1);
  float xs = sx1 + gx * (sx2 - sx1);
  float hm1 = (float)(H - 1);
  float y0f = fminf(fmaxf(floorf(ys), 0.0f), hm1);
  float x0f = fminf(fmaxf(floorf(xs), 0.0f), hm1);
  int y0 = (int)y0f, x0 = (int)x0f;
  int yb = min(y0 + 1, H - 1), xb = min(x0 + 1, H - 1);
  float wy = fminf(fmaxf(ys - y0f, 0.0f), 1.0f);
  float wx = fminf(fmaxf(xs - x0f, 0.0f), 1.0f);

  const float* f00 = feat + ((size_t)y0 * H + x0) * 256;
  const float* f01 = feat + ((size_t)y0 * H + xb) * 256;
  const float* f10 = feat + ((size_t)yb * H + x0) * 256;
  const float* f11 = feat + ((size_t)yb * H + xb) * 256;
  float v = f00[c] * (1.0f - wy) * (1.0f - wx) + f01[c] * (1.0f - wy) * wx +
            f10[c] * wy * (1.0f - wx)          + f11[c] * wy * wx;
  A1[(size_t)roi * K1 + (size_t)pos * 256 + c] = f2bf(v);
}

// -------------------------------------------------------------- WMMA GEMM
// C[M][N] = relu(A[M][K] * B[K][N] + bias), A bf16 row-major, B packed bf16.
// Wave computes a 32(M) x 64(N) tile: 2 A-tiles x 4 B-tiles = 8 accumulators.
// Block = 8 waves. tasks = (M/32)*(N/64).
__global__ __launch_bounds__(256) void gemm_bf16_kernel(
    const unsigned short* __restrict__ Ap, const unsigned short* __restrict__ Bpk,
    const float* __restrict__ bias, void* __restrict__ Cout,
    int M, int N, int K, int store_bf16) {
  int lane = threadIdx.x & 31;
  int wave = threadIdx.x >> 5;
  int task = blockIdx.x * 8 + wave;
  int n_supers = N >> 6;
  int m_super = task / n_supers;
  int n_super = task - m_super * n_supers;
  int m0 = m_super * 32;
  int n0 = n_super * 64;
  if (m0 >= M) return;

  int lhi = lane >> 4;     // 0|1 : which K half-group this lane owns
  int llo = lane & 15;     // row (A) / column (B) within tile

  v8f zero;
#pragma unroll
  for (int e = 0; e < 8; ++e) zero[e] = 0.0f;
  v8f acc[2][4];
#pragma unroll
  for (int a = 0; a < 2; ++a)
#pragma unroll
    for (int nn = 0; nn < 4; ++nn) acc[a][nn] = zero;

  // A: lane row = m0+llo (+16), halves: K [kb*32+lhi*8, +8) and +16
  const __bf16* arow0 = (const __bf16*)Ap + (size_t)(m0 + llo) * K + lhi * 8;
  const __bf16* arow1 = arow0 + (size_t)16 * K;
  // B packed: [kb][n][32]; lane reads halves [lhi*8, +8) and +16
  const __bf16* bbase = (const __bf16*)Bpk + lhi * 8;

  int nkb = K >> 5;
  for (int kb = 0; kb < nkb; ++kb) {
    v16bf a0, a1, b[4];
    const __bf16* ap0 = arow0 + kb * 32;
    ((float4*)&a0)[0] = *(const float4*)(const void*)(ap0);
    ((float4*)&a0)[1] = *(const float4*)(const void*)(ap0 + 16);
    const __bf16* ap1 = arow1 + kb * 32;
    ((float4*)&a1)[0] = *(const float4*)(const void*)(ap1);
    ((float4*)&a1)[1] = *(const float4*)(const void*)(ap1 + 16);
#pragma unroll
    for (int nn = 0; nn < 4; ++nn) {
      const __bf16* bp = bbase + ((size_t)kb * N + (n0 + nn * 16 + llo)) * 32;
      ((float4*)&b[nn])[0] = *(const float4*)(const void*)(bp);
      ((float4*)&b[nn])[1] = *(const float4*)(const void*)(bp + 16);
    }
    if (kb + 1 < nkb)   // gfx1250 global_prefetch_b8 of next B block
      __builtin_prefetch((const void*)(bbase + ((size_t)(kb + 1) * N + n0 + llo) * 32), 0, 0);
#pragma unroll
    for (int a = 0; a < 2; ++a)
#pragma unroll
      for (int nn = 0; nn < 4; ++nn)
        acc[a][nn] = __builtin_amdgcn_wmma_f32_16x16x32_bf16(
            false, (a == 0) ? a0 : a1, false, b[nn],
            (short)0, acc[a][nn], false, false);
  }

  // Epilogue: C/D layout — elem g, lane L: M = m0+a*16+(L>>4)*8+g, N = n0+nn*16+(L&15)
#pragma unroll
  for (int a = 0; a < 2; ++a) {
#pragma unroll
    for (int nn = 0; nn < 4; ++nn) {
      int n = n0 + nn * 16 + llo;
      float bv = bias[n];
      int mbase = m0 + a * 16 + lhi * 8;
#pragma unroll
      for (int g = 0; g < 8; ++g) {
        float v = acc[a][nn][g] + bv;
        v = v > 0.0f ? v : 0.0f;
        size_t idx = (size_t)(mbase + g) * N + n;
        if (store_bf16) ((unsigned short*)Cout)[idx] = f2bf(v);
        else            ((float*)Cout)[idx] = v;
      }
    }
  }
}

// --------------------------------------------------------- cls head (small)
__global__ __launch_bounds__(256) void cls_logits_kernel(
    const float* __restrict__ h2, const float* __restrict__ w,
    const float* __restrict__ b, float* __restrict__ logits) {
  int t = blockIdx.x * 256 + threadIdx.x;
  if (t >= NROI * NCLS) return;
  int r = t / NCLS, c = t - r * NCLS;
  const float* hr = h2 + (size_t)r * NH;
  float s = b[c];
  for (int k = 0; k < NH; ++k) s += hr[k] * w[k * NCLS + c];
  logits[t] = s;
}

__global__ void softmax_acc_kernel(const float* __restrict__ logits,
                                   float* __restrict__ out) {
  int r = blockIdx.x * 256 + threadIdx.x;
  if (r >= NROI) return;
  const float* l = logits + (size_t)r * NCLS;
  float mx = l[0];
  for (int c = 1; c < NCLS; ++c) mx = fmaxf(mx, l[c]);
  float s = 0.0f;
  for (int c = 0; c < NCLS; ++c) s += expf(l[c] - mx);
  float inv = 1.0f / (3.0f * s);
  for (int c = 0; c < NCLS; ++c) out[(size_t)r * NCLS + c] += expf(l[c] - mx) * inv;
}

// ------------------------------------- reg head + delta2bbox (wave per roi)
__global__ __launch_bounds__(256) void reg_bbox_kernel(
    const float* __restrict__ h2, const float* __restrict__ w,  // [1024][4]
    const float* __restrict__ b, const float* __restrict__ rin,
    float* __restrict__ rout) {
  int lane = threadIdx.x & 31;
  int r = blockIdx.x * 8 + (threadIdx.x >> 5);
  if (r >= NROI) return;
  const float* hr = h2 + (size_t)r * NH;
  float s0 = 0.f, s1 = 0.f, s2 = 0.f, s3 = 0.f;
  for (int k = lane; k < NH; k += 32) {
    float hv = hr[k];
    const float* wr = w + k * 4;
    s0 += hv * wr[0]; s1 += hv * wr[1]; s2 += hv * wr[2]; s3 += hv * wr[3];
  }
#pragma unroll
  for (int off = 16; off > 0; off >>= 1) {
    s0 += __shfl_xor(s0, off, 32);
    s1 += __shfl_xor(s1, off, 32);
    s2 += __shfl_xor(s2, off, 32);
    s3 += __shfl_xor(s3, off, 32);
  }
  if (lane == 0) {
    float d0 = (s0 + b[0]) * 0.1f, d1 = (s1 + b[1]) * 0.1f;
    float d2 = (s2 + b[2]) * 0.2f, d3 = (s3 + b[3]) * 0.2f;
    float y1 = rin[r * 4 + 0], x1 = rin[r * 4 + 1];
    float y2 = rin[r * 4 + 2], x2 = rin[r * 4 + 3];
    float h = y2 - y1, wd = x2 - x1;
    float cy = y1 + 0.5f * h + d0 * h;
    float cx = x1 + 0.5f * wd + d1 * wd;
    h  *= expf(d2);
    wd *= expf(d3);
    float b0 = fminf(fmaxf(cy - 0.5f * h, 0.0f), 1024.0f);
    float b1 = fminf(fmaxf(cx - 0.5f * wd, 0.0f), 1024.0f);
    float b2 = fminf(fmaxf(cy + 0.5f * h, 0.0f), 1024.0f);
    float b3 = fminf(fmaxf(cx + 0.5f * wd, 0.0f), 1024.0f);
    rout[r * 4 + 0] = b0; rout[r * 4 + 1] = b1;
    rout[r * 4 + 2] = b2; rout[r * 4 + 3] = b3;
  }
}

// ------------------------------------------------------------------ launch
static inline size_t align256(size_t x) { return (x + 255) & ~(size_t)255; }

extern "C" void kernel_launch(void* const* d_in, const int* in_sizes, int n_in,
                              void* d_out, int out_size, void* d_ws, size_t ws_size,
                              hipStream_t stream) {
  const float* P2    = (const float*)d_in[0];
  const float* P3    = (const float*)d_in[1];
  const float* P4    = (const float*)d_in[2];
  const float* P5    = (const float*)d_in[3];
  const float* rois  = (const float*)d_in[4];
  const float* fc1w  = (const float*)d_in[5];
  const float* fc1b  = (const float*)d_in[6];
  const float* fc2w  = (const float*)d_in[7];
  const float* fc2b  = (const float*)d_in[8];
  const float* clsw  = (const float*)d_in[9];
  const float* clsb  = (const float*)d_in[10];
  const float* regw  = (const float*)d_in[11];
  const float* regb  = (const float*)d_in[12];
  float* out = (float*)d_out;

  // workspace layout
  char* ws = (char*)d_ws;
  size_t off = 0;
  unsigned short* pW1 = (unsigned short*)(ws + off); off = align256(off + 3ull * K1 * NH * 2);
  unsigned short* pW2 = (unsigned short*)(ws + off); off = align256(off + 3ull * NH * NH * 2);
  unsigned short* A1  = (unsigned short*)(ws + off); off = align256(off + (size_t)MPAD * K1 * 2);
  unsigned short* A2  = (unsigned short*)(ws + off); off = align256(off + (size_t)MPAD * NH * 2);
  float*          h2  = (float*)(ws + off);          off = align256(off + (size_t)MPAD * NH * 4);
  float*          lg  = (float*)(ws + off);          off = align256(off + (size_t)NROI * NCLS * 4);
  float*          r1  = (float*)(ws + off);          off = align256(off + NROI * 4 * 4);
  float*          r2  = (float*)(ws + off);          off = align256(off + NROI * 4 * 4);
  (void)ws_size; (void)in_sizes; (void)n_in; (void)out_size;

  // init: zero output accumulator, zero A1 pad rows (1000..1023)
  zero_f32_kernel<<<(NROI * NCLS + 255) / 256, 256, 0, stream>>>(out, NROI * NCLS);
  {
    int padElems = (MPAD - NROI) * K1;
    zero_u16_kernel<<<(padElems + 255) / 256, 256, 0, stream>>>(
        A1 + (size_t)NROI * K1, padElems);
  }

  // pack weights to bf16 WMMA-B layout
  for (int j = 0; j < 3; ++j) {
    pack_w_kernel<<<(K1 / 32) * (NH / 256), 256, 0, stream>>>(
        fc1w + (size_t)j * K1 * NH, pW1 + (size_t)j * K1 * NH, K1, NH);
    pack_w_kernel<<<(NH / 32) * (NH / 256), 256, 0, stream>>>(
        fc2w + (size_t)j * NH * NH, pW2 + (size_t)j * NH * NH, NH, NH);
  }

  const int GEMM_BLOCKS = (MPAD / 32) * (NH / 64) / 8;  // 64

  // ---- stage 0: pooled -> fc1 -> fc2 -> reg -> r1 (probs unused)
  roi_align_kernel<<<NROI * 49, 256, 0, stream>>>(P2, P3, P4, P5, rois, A1);
  gemm_bf16_kernel<<<GEMM_BLOCKS, 256, 0, stream>>>(A1, pW1, fc1b, A2, MPAD, NH, K1, 1);
  gemm_bf16_kernel<<<GEMM_BLOCKS, 256, 0, stream>>>(A2, pW2, fc2b, h2, MPAD, NH, NH, 0);
  reg_bbox_kernel<<<(NROI + 7) / 8, 256, 0, stream>>>(h2, regw, regb, rois, r1);

  // ---- stage 1: r1 -> ... -> r2
  roi_align_kernel<<<NROI * 49, 256, 0, stream>>>(P2, P3, P4, P5, r1, A1);
  gemm_bf16_kernel<<<GEMM_BLOCKS, 256, 0, stream>>>(
      A1, pW1 + (size_t)1 * K1 * NH, fc1b + NH, A2, MPAD, NH, K1, 1);
  gemm_bf16_kernel<<<GEMM_BLOCKS, 256, 0, stream>>>(
      A2, pW2 + (size_t)1 * NH * NH, fc2b + NH, h2, MPAD, NH, NH, 0);
  reg_bbox_kernel<<<(NROI + 7) / 8, 256, 0, stream>>>(h2, regw + NH * 4, regb + 4, r1, r2);

  // ---- stage 2: pooled3 once, then heads j=0,1,2 -> softmax accumulate /3
  roi_align_kernel<<<NROI * 49, 256, 0, stream>>>(P2, P3, P4, P5, r2, A1);
  for (int j = 0; j < 3; ++j) {
    gemm_bf16_kernel<<<GEMM_BLOCKS, 256, 0, stream>>>(
        A1, pW1 + (size_t)j * K1 * NH, fc1b + (size_t)j * NH, A2, MPAD, NH, K1, 1);
    gemm_bf16_kernel<<<GEMM_BLOCKS, 256, 0, stream>>>(
        A2, pW2 + (size_t)j * NH * NH, fc2b + (size_t)j * NH, h2, MPAD, NH, NH, 0);
    cls_logits_kernel<<<(NROI * NCLS + 255) / 256, 256, 0, stream>>>(
        h2, clsw + (size_t)j * NH * NCLS, clsb + (size_t)j * NCLS, lg);
    softmax_acc_kernel<<<(NROI + 255) / 256, 256, 0, stream>>>(lg, out);
  }
}
